// Haptic2AudioRNN_22239340659241
// MI455X (gfx1250) — matrix operand order repaired and checked by reading, not compiled, not run on hardware
//
#include <hip/hip_runtime.h>

// Problem dims (compile-time constants)
#define RB 32    // batch
#define RS 1024  // sequence length
#define RI 128   // input features
#define RH 512   // hidden
#define RL 3     // layers
#define SCAN_NWG 2

typedef __attribute__((ext_vector_type(16))) __bf16 v16bf;
typedef __attribute__((ext_vector_type(8)))  float  v8f;

__device__ __forceinline__ v8f wmma_bf16(v16bf a, v16bf b, v8f c) {
  // D = A(16x32) * B(32x16) + C, f32 accumulate
  return __builtin_amdgcn_wmma_f32_16x16x32_bf16(false, a, false, b, (short)0, c,
                                                 false, false);
}

// ---- A-matrix (16x32 bf16) fragment loaders --------------------------------
// ISA layout: lanes 0-15 hold row M=lane with K in {0..7} U {16..23};
//             lanes 16-31 hold row M=lane-16 with K in {8..15} U {24..31}.
__device__ __forceinline__ v16bf load_a(const __bf16* __restrict__ base, int ld,
                                        int row, int kb, int hi) {
  const __bf16* p = base + (size_t)row * ld + kb + hi * 8;
  union { uint4 u; __bf16 h[8]; } q0, q1;
  q0.u = *(const uint4*)(p);
  q1.u = *(const uint4*)(p + 16);
  v16bf f;
#pragma unroll
  for (int j = 0; j < 8; ++j) { f[j] = q0.h[j]; f[j + 8] = q1.h[j]; }
  return f;
}

__device__ __forceinline__ v16bf load_a(const float* __restrict__ base, int ld,
                                        int row, int kb, int hi) {
  const float* p = base + (size_t)row * ld + kb + hi * 8;
  float4 c0a = *(const float4*)(p);
  float4 c0b = *(const float4*)(p + 4);
  float4 c1a = *(const float4*)(p + 16);
  float4 c1b = *(const float4*)(p + 20);
  v16bf f;
  f[0]  = (__bf16)c0a.x; f[1]  = (__bf16)c0a.y; f[2]  = (__bf16)c0a.z; f[3]  = (__bf16)c0a.w;
  f[4]  = (__bf16)c0b.x; f[5]  = (__bf16)c0b.y; f[6]  = (__bf16)c0b.z; f[7]  = (__bf16)c0b.w;
  f[8]  = (__bf16)c1a.x; f[9]  = (__bf16)c1a.y; f[10] = (__bf16)c1a.z; f[11] = (__bf16)c1a.w;
  f[12] = (__bf16)c1b.x; f[13] = (__bf16)c1b.y; f[14] = (__bf16)c1b.z; f[15] = (__bf16)c1b.w;
  return f;
}

// ---- B-matrix (32x16 bf16) fragment loader ---------------------------------
// ISA layout: lanes 0-15 hold col N=lane with K in {0..7} U {16..23};
//             lanes 16-31 hold col N=lane-16 with K in {8..15} U {24..31}.
// B[k][n] = W[n][k] with W row-major [N,K] -> contiguous along W's row.
__device__ __forceinline__ v16bf load_b(const float* __restrict__ w, int ld,
                                        int col, int kb, int hi) {
  const float* p = w + (size_t)col * ld + kb + hi * 8;
  float4 x0 = *(const float4*)(p);
  float4 x1 = *(const float4*)(p + 4);
  float4 x2 = *(const float4*)(p + 16);
  float4 x3 = *(const float4*)(p + 20);
  v16bf f;
  f[0]  = (__bf16)x0.x; f[1]  = (__bf16)x0.y; f[2]  = (__bf16)x0.z; f[3]  = (__bf16)x0.w;
  f[4]  = (__bf16)x1.x; f[5]  = (__bf16)x1.y; f[6]  = (__bf16)x1.z; f[7]  = (__bf16)x1.w;
  f[8]  = (__bf16)x2.x; f[9]  = (__bf16)x2.y; f[10] = (__bf16)x2.z; f[11] = (__bf16)x2.w;
  f[12] = (__bf16)x3.x; f[13] = (__bf16)x3.y; f[14] = (__bf16)x3.z; f[15] = (__bf16)x3.w;
  return f;
}

__device__ __forceinline__ float fast_tanh(float x) {
  float xc = fminf(fmaxf(x, -15.f), 15.f);   // avoid inf/inf
  float e  = __expf(2.f * xc);
  return (e - 1.f) / (e + 1.f);
}

// ============================================================================
// Projection GEMM: out[M,N] = in[M,K] @ W[N,K]^T + bias[N]   (f32 out)
// Block: 256 threads = 8 waves, wave tile 32(M) x 64(N); block tile 64 x 256.
// ============================================================================
template <typename Tin>
__launch_bounds__(256)
__global__ void proj_kernel(const Tin* __restrict__ in, const float* __restrict__ w,
                            const float* __restrict__ bias, float* __restrict__ out,
                            int M, int N, int K) {
  const int lane = threadIdx.x & 31;
  const int wave = threadIdx.x >> 5;
  const int m16  = lane & 15;
  const int hi   = lane >> 4;
  const int wm   = wave & 1;
  const int wn   = wave >> 1;
  const int mbase = blockIdx.x * 64 + wm * 32;
  const int nbase = blockIdx.y * 256 + wn * 64;

  v8f acc[2][4];
#pragma unroll
  for (int i = 0; i < 2; ++i)
#pragma unroll
    for (int j = 0; j < 4; ++j) acc[i][j] = (v8f){0.f,0.f,0.f,0.f,0.f,0.f,0.f,0.f};

  for (int kb = 0; kb < K; kb += 32) {
    v16bf a0 = load_a(in, K, mbase + m16, kb, hi);
    v16bf a1 = load_a(in, K, mbase + 16 + m16, kb, hi);
#pragma unroll
    for (int nt = 0; nt < 4; ++nt) {
      v16bf bf = load_b(w, K, nbase + nt * 16 + m16, kb, hi);
      acc[0][nt] = wmma_bf16(a0, bf, acc[0][nt]);
      acc[1][nt] = wmma_bf16(a1, bf, acc[1][nt]);
    }
  }

#pragma unroll
  for (int nt = 0; nt < 4; ++nt) {
    const int col = nbase + nt * 16 + m16;
    const float bv = bias[col];
#pragma unroll
    for (int r = 0; r < 8; ++r) {
      out[(size_t)(mbase + r + hi * 8) * N + col]      = acc[0][nt][r] + bv;
      out[(size_t)(mbase + 16 + r + hi * 8) * N + col] = acc[1][nt][r] + bv;
    }
  }
}

// ---- cross-workgroup barrier (monotonic counter, agent scope) --------------
__device__ __forceinline__ void gbarrier(unsigned* cnt, unsigned* bar) {
  *bar += 1;
  __syncthreads();
  __threadfence();  // publish this WG's h-state stores agent-wide
  if (threadIdx.x == 0) {
    const unsigned target = SCAN_NWG * (*bar);
    __hip_atomic_fetch_add(cnt, 1u, __ATOMIC_RELEASE, __HIP_MEMORY_SCOPE_AGENT);
    while (__hip_atomic_load(cnt, __ATOMIC_ACQUIRE, __HIP_MEMORY_SCOPE_AGENT) < target)
      __builtin_amdgcn_s_sleep(2);
  }
  __syncthreads();
  __threadfence();  // don't read stale h from caches
}

// ============================================================================
// Recurrent scan for one layer: h_t = tanh(pre[:,t,:] + h_{t-1} @ Whh^T + bhh)
// 2 persistent WGs x 512 threads (16 waves). Wave owns 16 output columns and
// keeps its Whh^T slice as 16 B-fragments in VGPRs (128 regs) for all steps.
// h state lives double-buffered in LDS (2x32KB); only the peer WG's 256-column
// half (16KB) crosses global per step, fetched with global_load_async_to_lds.
// ============================================================================
__launch_bounds__(512, 1)
__global__ void scan_kernel(const float* __restrict__ pre,   // [B,S,H] f32
                            const float* __restrict__ whh,   // [H,H]  f32
                            const float* __restrict__ bhh,   // [H]
                            const float* __restrict__ h0,    // [B,H]
                            __bf16* __restrict__ hbuf,       // [2,B,H] bf16 (global exch)
                            unsigned* __restrict__ cnt,      // zeroed before launch
                            float* __restrict__ hidden_out,  // [B,H] f32 (d_out slice)
                            __bf16* __restrict__ out_bf,     // [B,S,H] or null
                            float* __restrict__ out_f32) {   // [B,S,H] or null
  __shared__ __align__(16) __bf16 lh[2][RB * RH];  // 2 x 32KB double-buffered h

  const int lane = threadIdx.x & 31;
  const int wave = threadIdx.x >> 5;   // 0..15
  const int m16  = lane & 15;
  const int hi   = lane >> 4;
  const int wg   = blockIdx.x;         // 0..1
  const int ncol = wg * 256 + wave * 16 + m16;   // this lane's output column
  const int obase = (1 - wg) * 256;              // peer WG's column base

  // Pin W_hh^T B-fragments in registers: 16 K-tiles x 8 VGPRs = 128 VGPRs.
  v16bf bw[16];
#pragma unroll
  for (int kt = 0; kt < 16; ++kt) bw[kt] = load_b(whh, RH, ncol, kt * 32, hi);

  const float bias = bhh[ncol];

  // Init h-state (own 256-column half) into LDS buf0 + global buf0.
  for (int i = threadIdx.x; i < RB * 256; i += blockDim.x) {
    const int b = i >> 8;
    const int c = (i & 255) + wg * 256;
    const __bf16 v = (__bf16)h0[b * RH + c];
    lh[0][b * RH + c] = v;
    hbuf[b * RH + c]  = v;
  }

  unsigned bar = 0;
  gbarrier(cnt, &bar);

  for (int t = 0; t < RS; ++t) {
    // ---- stage peer WG's half of h_t from global into LDS (async DMA path)
    {
      const __bf16* gsrc = hbuf + (size_t)(t & 1) * (RB * RH);
#pragma unroll
      for (int u = threadIdx.x; u < RB * 32; u += 512) {  // 1024 x 16B units
        const int row = u >> 5;
        const int off = (u & 31) * 8;            // bf16 elems within the half
        const int idx = row * RH + obase + off;
        const unsigned lds_addr = (unsigned)(unsigned long long)(void*)&lh[t & 1][idx];
        const unsigned long long gaddr = (unsigned long long)(gsrc + idx);
        asm volatile("global_load_async_to_lds_b128 %0, %1, off"
                     :: "v"(lds_addr), "v"(gaddr) : "memory");
      }
      asm volatile("s_wait_asynccnt 0" ::: "memory");
      __syncthreads();
    }
    const __bf16* hlds = &lh[t & 1][0];
    __bf16* hwrite_l   = &lh[(t + 1) & 1][0];
    __bf16* hwrite_g   = hbuf + (size_t)((t + 1) & 1) * (RB * RH);

    // C init = pre[:,t,ncol] + b_hh
    v8f c0, c1;
#pragma unroll
    for (int r = 0; r < 8; ++r) {
      const int b0 = r + hi * 8;
      c0[r] = pre[((size_t)b0 * RS + t) * RH + ncol] + bias;
      c1[r] = pre[((size_t)(b0 + 16) * RS + t) * RH + ncol] + bias;
    }
    // Prefetch next step's pre rows while this step computes / spins.
    if (t + 1 < RS) {
#pragma unroll
      for (int r = 0; r < 8; ++r) {
        const int b0 = r + hi * 8;
        __builtin_prefetch(&pre[((size_t)b0 * RS + t + 1) * RH + ncol], 0, 1);
        __builtin_prefetch(&pre[((size_t)(b0 + 16) * RS + t + 1) * RH + ncol], 0, 1);
      }
    }

    // h @ Whh^T over K=512: 16 k-tiles, 2 M-tiles -> 32 WMMAs per wave,
    // A-fragments from LDS (ds_load_b128).
#pragma unroll
    for (int kt = 0; kt < 16; ++kt) {
      v16bf a0 = load_a(hlds, RH, m16, kt * 32, hi);
      v16bf a1 = load_a(hlds, RH, 16 + m16, kt * 32, hi);
      c0 = wmma_bf16(a0, bw[kt], c0);
      c1 = wmma_bf16(a1, bw[kt], c1);
    }

#pragma unroll
    for (int r = 0; r < 8; ++r) {
      const int b0 = r + hi * 8;
      const int b1 = b0 + 16;
      const float h0n = fast_tanh(c0[r]);
      const float h1n = fast_tanh(c1[r]);
      const __bf16 h0b = (__bf16)h0n;
      const __bf16 h1b = (__bf16)h1n;
      hwrite_l[b0 * RH + ncol] = h0b;   // own half -> LDS (next step's A)
      hwrite_l[b1 * RH + ncol] = h1b;
      hwrite_g[b0 * RH + ncol] = h0b;   // own half -> global (for peer WG)
      hwrite_g[b1 * RH + ncol] = h1b;
      if (out_f32) {
        out_f32[((size_t)b0 * RS + t) * RH + ncol] = h0n;
        out_f32[((size_t)b1 * RS + t) * RH + ncol] = h1n;
      } else {
        out_bf[((size_t)b0 * RS + t) * RH + ncol] = h0b;
        out_bf[((size_t)b1 * RS + t) * RH + ncol] = h1b;
      }
      if (t == RS - 1) {
        hidden_out[b0 * RH + ncol] = h0n;
        hidden_out[b1 * RH + ncol] = h1n;
      }
    }
    gbarrier(cnt, &bar);  // h_{t+1} published; double buffers remove WAR hazards
  }
}

// ============================================================================
extern "C" void kernel_launch(void* const* d_in, const int* in_sizes, int n_in,
                              void* d_out, int out_size, void* d_ws, size_t ws_size,
                              hipStream_t stream) {
  (void)in_sizes; (void)n_in; (void)out_size; (void)ws_size;
  const float* x     = (const float*)d_in[0];   // [B,S,I]
  const float* h0    = (const float*)d_in[1];   // [L,B,H]
  const float* w_ih0 = (const float*)d_in[2];   // [H,I]
  const float* w_ihs = (const float*)d_in[3];   // [L-1,H,H]
  const float* w_hhs = (const float*)d_in[4];   // [L,H,H]
  const float* b_ihs = (const float*)d_in[5];   // [L,H]
  const float* b_hhs = (const float*)d_in[6];   // [L,H]
  float* out = (float*)d_out;

  // Workspace layout (needs ~96.1 MB):
  //   pre  : f32 [B,S,H]        64 MB
  //   outb : bf16 [B,S,H]       32 MB  (inter-layer activations)
  //   hbuf : bf16 [2,B,H]       64 KB  (double-buffered h exchange)
  //   cnts : 3 barrier counters (128B apart)
  char* ws = (char*)d_ws;
  float*    pre  = (float*)ws;
  __bf16*   outb = (__bf16*)(ws + (size_t)RB * RS * RH * 4);
  __bf16*   hbuf = (__bf16*)(ws + (size_t)RB * RS * RH * 6);
  unsigned* cnts = (unsigned*)(ws + (size_t)RB * RS * RH * 6 + (size_t)2 * RB * RH * 2);

  hipMemsetAsync(cnts, 0, 3 * 128, stream);  // deterministic barrier state per call

  float* hidden  = out;                          // [L,B,H]
  float* outTail = out + (size_t)RL * RB * RH;   // [B*S,H]

  const int M = RB * RS;
  dim3 pgrid(M / 64, RH / 256);

  // ---- layer 0
  proj_kernel<float><<<pgrid, 256, 0, stream>>>(x, w_ih0, b_ihs, pre, M, RH, RI);
  scan_kernel<<<SCAN_NWG, 512, 0, stream>>>(pre, w_hhs, b_hhs, h0, hbuf,
                                            (unsigned*)((char*)cnts + 0),
                                            hidden, outb, nullptr);
  // ---- layer 1
  proj_kernel<__bf16><<<pgrid, 256, 0, stream>>>(outb, w_ihs, b_ihs + RH, pre, M, RH, RH);
  scan_kernel<<<SCAN_NWG, 512, 0, stream>>>(pre, w_hhs + (size_t)RH * RH, b_hhs + RH,
                                            h0 + (size_t)RB * RH, hbuf,
                                            (unsigned*)((char*)cnts + 128),
                                            hidden + (size_t)RB * RH, outb, nullptr);
  // ---- layer 2 (f32 outputs straight into d_out tail)
  proj_kernel<__bf16><<<pgrid, 256, 0, stream>>>(outb, w_ihs + (size_t)RH * RH,
                                                 b_ihs + 2 * RH, pre, M, RH, RH);
  scan_kernel<<<SCAN_NWG, 512, 0, stream>>>(pre, w_hhs + (size_t)2 * RH * RH,
                                            b_hhs + 2 * RH, h0 + (size_t)2 * RB * RH, hbuf,
                                            (unsigned*)((char*)cnts + 256),
                                            hidden + (size_t)2 * RB * RH, nullptr, outTail);
}
